// ContrastiveFeatureTransformer_42399917146500
// MI455X (gfx1250) — matrix operand order repaired and checked by reading, not compiled, not run on hardware
//
#include <hip/hip_runtime.h>
#include <hip/hip_bf16.h>

// ---- problem constants (from reference setup_inputs) ----
#define BATCH 8
#define CIN   256
#define COUT  128
#define HWSZ  3600            // 60*60
#define NPIX  (BATCH*HWSZ)    // 28800
#define MT    (NPIX/16)       // 1800 row tiles
#define NT    (HWSZ/16)       // 225 col tiles per batch
#define INV_T (1.0f/0.07f)
#define BN_EPS 1e-5f
#define NORM_EPS 1e-12f
#define NCE_BLKS 29           // ceil(225 / 8) i-tile groups per batch

typedef __attribute__((ext_vector_type(16))) _Float16 v16h;
typedef __attribute__((ext_vector_type(8)))  float    v8f;

// CDNA5 async global->LDS copy (ASYNCcnt-tracked). VDST = 32-bit wave-relative
// LDS byte address (low 32 bits of flat LDS pointer), VADDR = 64-bit global.
__device__ __forceinline__ void async_copy_b128(uint32_t lds_addr, const void* gaddr) {
  asm volatile("global_load_async_to_lds_b128 %0, %1, off"
               :: "v"(lds_addr), "v"(gaddr) : "memory");
}
__device__ __forceinline__ void wait_async_le1() {
  asm volatile("s_wait_asynccnt 0x1" ::: "memory");
}
__device__ __forceinline__ void wait_async_0() {
  asm volatile("s_wait_asynccnt 0x0" ::: "memory");
}

// ============================================================
// init: zero the small accumulators in workspace
// ============================================================
__global__ void k_init(float* gsum, float* gss, float* acc) {
  int i = threadIdx.x;
  if (i < 2*COUT) { gsum[i] = 0.f; gss[i] = 0.f; }
  if (i < 2) acc[i] = 0.f;
}

// ============================================================
// GEMM1: y[p][n] = relu(x)[p][:] . conv_w[n][:] + conv_b[n]
//   x is [B, CIN, H, W] (channel-major), y is [NPIX, COUT] row-major, f32.
//   8 waves per block: wave w owns n-tile w (COUT = 8*16).
// ============================================================
__global__ __launch_bounds__(256)
void k_gemm1(const float* __restrict__ x0, const float* __restrict__ x1,
             const float* __restrict__ w1, const float* __restrict__ b1,
             float* __restrict__ y) {
  const int br   = blockIdx.y;
  const float* x = br ? x1 : x0;
  float* yb      = y + (size_t)br * NPIX * COUT;
  const int mt   = blockIdx.x;
  const int wave = threadIdx.x >> 5;
  const int lane = threadIdx.x & 31;
  const int hf   = lane >> 4;      // which 16-lane half
  const int l16  = lane & 15;

  // A-matrix row = pixel (16-bit A 16x32 layout: m = lane&15)
  const int pixA = mt*16 + l16;
  const int bA   = pixA / HWSZ;
  const int hwA  = pixA % HWSZ;
  const float* xA = x + (size_t)bA * CIN * HWSZ + hwA;

  // B-matrix column = output channel n (n = lane&15 within n-tile)
  const int n = wave*16 + l16;
  const float* wrow = w1 + (size_t)n * CIN;

  v8f c = {};
  #pragma unroll
  for (int kt = 0; kt < CIN/32; ++kt) {
    v16h a, bb;
    const int akb = kt*32 + hf*8;
    #pragma unroll
    for (int e = 0; e < 16; ++e) {       // A: k = akb + (e&7) (+16 for upper 8 elems)
      int k = akb + (e & 7) + ((e >= 8) ? 16 : 0);
      a[e] = (_Float16)fmaxf(xA[(size_t)k * HWSZ], 0.0f);   // fused prepend-relu
    }
    const int bkb = kt*32 + hf*8;
    #pragma unroll
    for (int e = 0; e < 16; ++e)         // B: k = bkb + (e&7) (+16 for upper 8 elems)
      bb[e] = (_Float16)wrow[bkb + (e & 7) + ((e >= 8) ? 16 : 0)];
    c = __builtin_amdgcn_wmma_f32_16x16x32_f16(false, a, false, bb,
                                               (short)0, c, false, false);
  }
  // C layout: VGPR r -> row = r + 8*hf, col = l16
  const float bias = b1[n];
  #pragma unroll
  for (int r = 0; r < 8; ++r)
    yb[(size_t)(mt*16 + hf*8 + r) * COUT + n] = c[r] + bias;
}

// ============================================================
// BN stats: per-channel sum / sumsq over 28800 rows (coalesced: tid = channel)
// ============================================================
__global__ __launch_bounds__(128)
void k_stats(const float* __restrict__ y, float* __restrict__ gsum,
             float* __restrict__ gss) {
  const int br = blockIdx.y;
  const float* yb = y + (size_t)br * NPIX * COUT;
  const int ch = threadIdx.x;
  const int r0 = blockIdx.x * 128;         // 225 blocks * 128 rows = 28800
  float s = 0.f, ss = 0.f;
  for (int r = 0; r < 128; ++r) {
    float v = yb[(size_t)(r0 + r) * COUT + ch];
    s += v; ss += v * v;
  }
  atomicAdd(&gsum[br*COUT + ch], s);
  atomicAdd(&gss [br*COUT + ch], ss);
}

// ============================================================
// fold BN(training stats) + gamma/beta into per-channel (scale, shift)
// ============================================================
__global__ void k_bnfold(const float* __restrict__ gsum, const float* __restrict__ gss,
                         const float* __restrict__ gamma, const float* __restrict__ beta,
                         float2* __restrict__ sc) {
  int i = threadIdx.x;                 // 0..255 covers 2 branches x 128 channels
  if (i >= 2*COUT) return;
  int ch = i & (COUT - 1);
  float mean  = gsum[i] * (1.0f / NPIX);
  float var   = gss[i]  * (1.0f / NPIX) - mean * mean;
  float scale = gamma[ch] * rsqrtf(var + BN_EPS);
  sc[i] = make_float2(scale, beta[ch] - mean * scale);
}

// ============================================================
// GEMM2 + L2 normalize:
//   z[p][n] = relu(y[p][:]*scale+shift) . lin_w[n][:] + lin_b[n]
//   feat[p][:] = z / max(||z||, 1e-12)   stored as f16 [NPIX][COUT]
// ============================================================
__global__ __launch_bounds__(256)
void k_gemm2(const float* __restrict__ y, const float2* __restrict__ sc,
             const float* __restrict__ w2, const float* __restrict__ b2,
             _Float16* __restrict__ feat) {
  __shared__ float zt[16][COUT];
  __shared__ float rn[16];
  const int br = blockIdx.y;
  const float*  yb  = y    + (size_t)br * NPIX * COUT;
  _Float16*     fb  = feat + (size_t)br * NPIX * COUT;
  const float2* scb = sc + br * COUT;
  const int mt   = blockIdx.x;
  const int wave = threadIdx.x >> 5;
  const int lane = threadIdx.x & 31;
  const int hf   = lane >> 4;
  const int l16  = lane & 15;

  const float* yrow = yb + (size_t)(mt*16 + l16) * COUT;
  const int n = wave*16 + l16;
  const float* wrow = w2 + (size_t)n * COUT;

  v8f c = {};
  #pragma unroll
  for (int kt = 0; kt < COUT/32; ++kt) {
    v16h a, bb;
    const int akb = kt*32 + hf*8;
    #pragma unroll
    for (int e = 0; e < 16; ++e) {
      int k = akb + (e & 7) + ((e >= 8) ? 16 : 0);
      float2 s = scb[k];
      a[e] = (_Float16)fmaxf(yrow[k] * s.x + s.y, 0.0f);   // BN + relu fused
    }
    const int bkb = kt*32 + hf*8;
    #pragma unroll
    for (int e = 0; e < 16; ++e) bb[e] = (_Float16)wrow[bkb + (e & 7) + ((e >= 8) ? 16 : 0)];
    c = __builtin_amdgcn_wmma_f32_16x16x32_f16(false, a, false, bb,
                                               (short)0, c, false, false);
  }
  const float bias = b2[n];
  #pragma unroll
  for (int r = 0; r < 8; ++r) zt[hf*8 + r][n] = c[r] + bias;
  __syncthreads();
  if (threadIdx.x < 16) {
    float ssq = 0.f;
    #pragma unroll 4
    for (int k = 0; k < COUT; ++k) { float v = zt[threadIdx.x][k]; ssq += v * v; }
    rn[threadIdx.x] = 1.0f / fmaxf(sqrtf(ssq), NORM_EPS);
  }
  __syncthreads();
  {
    int row = threadIdx.x >> 4;
    int c0  = (threadIdx.x & 15) * 8;
    float r = rn[row];
    #pragma unroll
    for (int j = 0; j < 8; ++j)
      fb[(size_t)(mt*16 + row) * COUT + c0 + j] = (_Float16)(zt[row][c0 + j] * r);
  }
}

// ============================================================
// pos: sum_i dot(o_i, t_i)/T  -> acc[1]
// ============================================================
__global__ __launch_bounds__(256)
void k_pos(const _Float16* __restrict__ fo, const _Float16* __restrict__ ft,
           float* __restrict__ acc) {
  __shared__ float red[256];
  int r = blockIdx.x * 256 + threadIdx.x;
  float p = 0.f;
  if (r < NPIX) {
    const _Float16* a = fo + (size_t)r * COUT;
    const _Float16* b = ft + (size_t)r * COUT;
    float d = 0.f;
    #pragma unroll 8
    for (int k = 0; k < COUT; ++k) d += (float)a[k] * (float)b[k];
    p = d * INV_T;
  }
  red[threadIdx.x] = p;
  __syncthreads();
  for (int s = 128; s > 0; s >>= 1) {
    if (threadIdx.x < s) red[threadIdx.x] += red[threadIdx.x + s];
    __syncthreads();
  }
  if (threadIdx.x == 0) atomicAdd(&acc[1], red[0]);
}

// ============================================================
// dense InfoNCE denominator, fused flash-style, LDS-blocked:
//   block = 8 waves, wave w owns i-tile (blockIdx.x*8 + w) of batch b.
//   All 8 waves share one stream of 225 t-tiles: each 16x128 f16 tile (4KB)
//   is async-copied global->LDS (double-buffered, ASYNCcnt), then every wave
//   reads B fragments from LDS (ds_load) -> 4 wmma -> rowsum += exp(S/T).
//   L2 arithmetic intensity: 8 waves * 16 FLOP/B = 128 FLOP/B.
//   No max-shift needed: |S/T| <= 1/0.07 = 14.3, sum <= 5.8e9 << f32 max.
// ============================================================
__global__ __launch_bounds__(256)
void k_nce(const _Float16* __restrict__ fo, const _Float16* __restrict__ ft,
           float* __restrict__ acc) {
  __shared__ _Float16 tile[2][16][COUT];          // 2 x 4KB double buffer
  const int b    = blockIdx.y;                    // batch 0..7
  const int wave = threadIdx.x >> 5;
  const int lane = threadIdx.x & 31;
  const int hf   = lane >> 4;
  const int l16  = lane & 15;
  const int it   = blockIdx.x * 8 + wave;         // this wave's i-tile
  const bool active = (it < NT);
  const int itc  = active ? it : (NT - 1);        // clamp for safe addressing

  // cooperative tile fill: thread t copies 16B: row = t/16, cols = (t%16)*8..
  const int fr  = threadIdx.x >> 4;
  const int fc  = (threadIdx.x & 15) * 8;
  const uint32_t lds0 = (uint32_t)(uintptr_t)&tile[0][fr][fc];
  const uint32_t lds1 = (uint32_t)(uintptr_t)&tile[1][fr][fc];
  const _Float16* gsrc = ft + (size_t)(b*HWSZ + fr) * COUT + fc;

  // A fragments for this wave's i-tile (registers for the whole j-loop)
  const _Float16* orow = fo + (size_t)(b*HWSZ + itc*16 + l16) * COUT;
  v16h a[4];
  #pragma unroll
  for (int kt = 0; kt < 4; ++kt) {
    const int akb = kt*32 + hf*8;
    #pragma unroll
    for (int e = 0; e < 16; ++e)
      a[kt][e] = orow[akb + (e & 7) + ((e >= 8) ? 16 : 0)];
  }

  // prime buffer 0 with j=0
  async_copy_b128(lds0, gsrc);

  float accr[8] = {0.f,0.f,0.f,0.f,0.f,0.f,0.f,0.f};
  for (int j = 0; j < NT; ++j) {
    const int cur = j & 1;
    if (j + 1 < NT) {                 // prefetch next tile into other buffer
      async_copy_b128(cur ? lds0 : lds1, gsrc + (size_t)(j + 1) * 16 * COUT);
      wait_async_le1();               // oldest (current buffer) has landed
    } else {
      wait_async_0();
    }
    __syncthreads();                  // all waves' fills of buf[cur] visible

    // B fragments from LDS: column n = l16 (t row), k blocked by half
    v8f c = {};
    #pragma unroll
    for (int kt = 0; kt < 4; ++kt) {
      v16h bb;
      const int bkb = kt*32 + hf*8;
      #pragma unroll
      for (int e = 0; e < 16; ++e) bb[e] = tile[cur][l16][bkb + (e & 7) + ((e >= 8) ? 16 : 0)];
      c = __builtin_amdgcn_wmma_f32_16x16x32_f16(false, a[kt], false, bb,
                                                 (short)0, c, false, false);
    }
    #pragma unroll
    for (int r = 0; r < 8; ++r) accr[r] += __expf(c[r] * INV_T);

    __syncthreads();                  // done reading buf[cur] before refill
  }

  // reduce exp-sums over the 16 columns held by each wave32 half
  #pragma unroll
  for (int r = 0; r < 8; ++r) {
    float v = accr[r];
    v += __shfl_xor(v, 1, 32);
    v += __shfl_xor(v, 2, 32);
    v += __shfl_xor(v, 4, 32);
    v += __shfl_xor(v, 8, 32);
    accr[r] = v;                      // lane0: rows 0..7, lane16: rows 8..15
  }
  if (active && l16 == 0) {
    float s = 0.f;
    #pragma unroll
    for (int r = 0; r < 8; ++r) s += __logf(accr[r]);   // lse (no max shift)
    atomicAdd(&acc[0], s);
  }
}

// ============================================================
// final: loss = (sum_lse - sum_pos) / NPIX
// ============================================================
__global__ void k_final(const float* __restrict__ acc, float* __restrict__ out) {
  out[0] = (acc[0] - acc[1]) * (1.0f / NPIX);
}

// ============================================================
extern "C" void kernel_launch(void* const* d_in, const int* in_sizes, int n_in,
                              void* d_out, int out_size, void* d_ws, size_t ws_size,
                              hipStream_t stream) {
  const float* x_orig = (const float*)d_in[0];
  const float* x_trans= (const float*)d_in[1];
  const float* conv_w = (const float*)d_in[2];
  const float* conv_b = (const float*)d_in[3];
  const float* gamma  = (const float*)d_in[4];
  const float* beta   = (const float*)d_in[5];
  const float* lin_w  = (const float*)d_in[6];
  const float* lin_b  = (const float*)d_in[7];
  float* out = (float*)d_out;

  // workspace layout
  char* base = (char*)d_ws;
  const size_t Y_BYTES    = (size_t)2 * NPIX * COUT * sizeof(float);    // 29.49 MB
  const size_t FEAT_BYTES = (size_t)2 * NPIX * COUT * sizeof(_Float16); // 14.75 MB
  float*     y    = (float*)base;
  _Float16*  feat = (_Float16*)(base + Y_BYTES);
  char* tail = base + Y_BYTES + FEAT_BYTES;
  float2* sc   = (float2*)tail;                 tail += 2*COUT*sizeof(float2);
  float*  gsum = (float*)tail;                  tail += 2*COUT*sizeof(float);
  float*  gss  = (float*)tail;                  tail += 2*COUT*sizeof(float);
  float*  acc  = (float*)tail;                  // [0]=sum_lse, [1]=sum_pos

  _Float16* feat_o = feat;
  _Float16* feat_t = feat + (size_t)NPIX * COUT;

  k_init  <<<1, 256, 0, stream>>>(gsum, gss, acc);
  k_gemm1 <<<dim3(MT, 2), 256, 0, stream>>>(x_orig, x_trans, conv_w, conv_b, y);
  k_stats <<<dim3(NT, 2), 128, 0, stream>>>(y, gsum, gss);
  k_bnfold<<<1, 256, 0, stream>>>(gsum, gss, gamma, beta, sc);
  k_gemm2 <<<dim3(MT, 2), 256, 0, stream>>>(y, sc, lin_w, lin_b, feat);
  k_pos   <<<(NPIX + 255) / 256, 256, 0, stream>>>(feat_o, feat_t, acc);
  k_nce   <<<dim3(NCE_BLKS, BATCH), 256, 0, stream>>>(feat_o, feat_t, acc);
  k_final <<<1, 1, 0, stream>>>(acc, out);
}
